// LHmmLm_48232482734151
// MI455X (gfx1250) — matrix operand order repaired and checked by reading, not compiled, not run on hardware
//
#include <hip/hip_runtime.h>
#include <hip/hip_bf16.h>
#include <stdint.h>

typedef __bf16 bf16;
typedef __attribute__((ext_vector_type(16))) __bf16 v16bf;
typedef __attribute__((ext_vector_type(8)))  float  v8f;

#define HN 256
#define CN 1024
#define TSTEPS 512
#define NB 8
#define VN 32000
#define NT (NB * TSTEPS) /* 4096 */
#define SCAN_WGS 16
#define LDP (CN + 8) /* padded row length for LDS tiles */
#define NEG_INF (-__builtin_huge_valf())

// ---------- WMMA fragment helpers (CDNA5 16x16x32 bf16 layouts) ----------
// A (16x32 bf16): lane holds row (lane&15); 16 contiguous halves at
// k0 + ((lane&16)?16:0).
__device__ __forceinline__ v16bf ld_a(const bf16* p) {
  v16bf f;
#pragma unroll
  for (int e = 0; e < 16; ++e) f[e] = p[e];
  return f;
}
// B (32x16 bf16): lane holds col (lane&15); 16 contiguous halves at
// k0 + ((lane&16)?16:0).
__device__ __forceinline__ v16bf ld_b(const bf16* p) {
  v16bf f;
#pragma unroll
  for (int e = 0; e < 16; ++e) f[e] = p[e];
  return f;
}
__device__ __forceinline__ v16bf ld_b_f32(const float* p) {
  v16bf f;
#pragma unroll
  for (int e = 0; e < 16; ++e) f[e] = (bf16)p[e];
  return f;
}
__device__ __forceinline__ v8f wmma_bf16(v16bf a, v16bf b, v8f c) {
  return __builtin_amdgcn_wmma_f32_16x16x32_bf16(false, a, false, b, (short)0, c,
                                                 false, false);
}

// ---------- CDNA5 async memory->LDS DMA (ASYNCcnt path) ----------
__device__ __forceinline__ void async_g2l_b128(unsigned lds_byte_addr,
                                               const void* gaddr) {
  asm volatile("global_load_async_to_lds_b128 %0, %1, off"
               :: "v"(lds_byte_addr), "v"(gaddr)
               : "memory");
}
__device__ __forceinline__ void wait_async0() {
  asm volatile("s_wait_asynccnt 0x0" ::: "memory");
}

// ---------- init: zero the grid-barrier counter ----------
__global__ void k_init(unsigned* c) { *c = 0u; }

// ---------- convert terminal_emb to bf16 once ----------
__global__ void k_cvt(const float* __restrict__ src, bf16* __restrict__ dst) {
  const int idx = blockIdx.x * 256 + threadIdx.x;
  dst[idx] = (bf16)src[idx];
}

// ---------- start vector: MLP + log_softmax vs next_state_emb ----------
__global__ void k_start(const float* __restrict__ emb, const float* __restrict__ W,
                        const float* __restrict__ b, const float* __restrict__ rW1,
                        const float* __restrict__ rb1, const float* __restrict__ rW2,
                        const float* __restrict__ rb2, const float* __restrict__ next_emb,
                        float* __restrict__ startv) {
  __shared__ float fx[HN], tb[HN], lg[CN], red[HN];
  const int tid = threadIdx.x;
  float s = b[tid];
  for (int k = 0; k < HN; ++k) s += W[tid * HN + k] * emb[k];
  fx[tid] = s;
  __syncthreads();
  for (int i = 0; i < 2; ++i) {
    const float* W1 = rW1 + (size_t)i * HN * HN;
    const float* W2 = rW2 + (size_t)i * HN * HN;
    float t1 = rb1[i * HN + tid];
    for (int k = 0; k < HN; ++k) t1 += W1[tid * HN + k] * fx[k];
    tb[tid] = fmaxf(t1, 0.f);
    __syncthreads();
    float t2 = rb2[i * HN + tid];
    for (int k = 0; k < HN; ++k) t2 += W2[tid * HN + k] * tb[k];
    float nv = fx[tid] + fmaxf(t2, 0.f);
    __syncthreads();
    fx[tid] = nv;
    __syncthreads();
  }
  for (int c = tid; c < CN; c += HN) {
    float l = 0.f;
    for (int k = 0; k < HN; ++k) l += fx[k] * next_emb[(size_t)c * HN + k];
    lg[c] = l;
  }
  __syncthreads();
  float mx = fmaxf(fmaxf(lg[tid], lg[tid + 256]), fmaxf(lg[tid + 512], lg[tid + 768]));
  red[tid] = mx;
  __syncthreads();
  for (int off = 128; off > 0; off >>= 1) {
    if (tid < off) red[tid] = fmaxf(red[tid], red[tid + off]);
    __syncthreads();
  }
  float M = red[0];
  __syncthreads();
  float se = __expf(lg[tid] - M) + __expf(lg[tid + 256] - M) +
             __expf(lg[tid + 512] - M) + __expf(lg[tid + 768] - M);
  red[tid] = se;
  __syncthreads();
  for (int off = 128; off > 0; off >>= 1) {
    if (tid < off) red[tid] += red[tid + off];
    __syncthreads();
  }
  float lse = M + __logf(red[0]);
  __syncthreads();
  for (int c = tid; c < CN; c += HN) startv[c] = lg[c] - lse;
}

// ---------- terminal MLP: ft = ResLayers(preterminal_emb), stored bf16 ----------
__global__ void k_termmlp(const float* __restrict__ pre, const float* __restrict__ rW1,
                          const float* __restrict__ rb1, const float* __restrict__ rW2,
                          const float* __restrict__ rb2, bf16* __restrict__ ft_bf) {
  __shared__ float fx[HN], tb[HN];
  const int c = blockIdx.x, tid = threadIdx.x;
  fx[tid] = pre[(size_t)c * HN + tid];
  __syncthreads();
  for (int i = 0; i < 2; ++i) {
    const float* W1 = rW1 + (size_t)i * HN * HN;
    const float* W2 = rW2 + (size_t)i * HN * HN;
    float t1 = rb1[i * HN + tid];
    for (int k = 0; k < HN; ++k) t1 += W1[tid * HN + k] * fx[k];
    tb[tid] = fmaxf(t1, 0.f);
    __syncthreads();
    float t2 = rb2[i * HN + tid];
    for (int k = 0; k < HN; ++k) t2 += W2[tid * HN + k] * tb[k];
    float nv = fx[tid] + fmaxf(t2, 0.f);
    __syncthreads();
    fx[tid] = nv;
    __syncthreads();
  }
  ft_bf[(size_t)c * HN + tid] = (bf16)fx[tid];
}

// ---------- transition: TexpT[j][i] = exp(log_softmax_j(state@next^T)) ----------
__global__ void k_transition(const float* __restrict__ state_emb,
                             const float* __restrict__ next_emb,
                             bf16* __restrict__ TexpT) {
  __shared__ bf16 As[16][HN];
  __shared__ float rowred[16][128];
  __shared__ float rowstat[16];
  const int tid = threadIdx.x, lane = tid & 31, wave = tid >> 5;
  const int i0 = blockIdx.x * 16;
  for (int idx = tid; idx < 16 * HN; idx += 256)
    As[idx >> 8][idx & 255] = (bf16)state_emb[(size_t)(i0 + (idx >> 8)) * HN + (idx & 255)];
  __syncthreads();
  const int arow = lane & 15;
  const int aoff = (lane & 16) ? 16 : 0;
  const int bcol = lane & 15;
  const int boff = (lane & 16) ? 16 : 0;
  const int rbase = (lane & 16) ? 8 : 0;
  v8f acc[8];
#pragma unroll
  for (int u = 0; u < 8; ++u) {
    const int j0 = (wave * 8 + u) * 16;
    v8f c = {};
    for (int k0 = 0; k0 < HN; k0 += 32) {
      v16bf a = ld_a(&As[arow][k0 + aoff]);
      v16bf b = ld_b_f32(next_emb + (size_t)(j0 + bcol) * HN + k0 + boff);
      c = wmma_bf16(a, b, c);
    }
    acc[u] = c;
  }
  const int slot = wave * 16 + (lane & 15);
#pragma unroll
  for (int q = 0; q < 8; ++q) {
    float M = NEG_INF;
#pragma unroll
    for (int u = 0; u < 8; ++u) M = fmaxf(M, acc[u][q]);
    rowred[rbase + q][slot] = M;
  }
  __syncthreads();
  if (tid < 16) {
    float M = NEG_INF;
    for (int s = 0; s < 128; ++s) M = fmaxf(M, rowred[tid][s]);
    rowstat[tid] = M;
  }
  __syncthreads();
#pragma unroll
  for (int q = 0; q < 8; ++q) {
    float M = rowstat[rbase + q];
    float S = 0.f;
#pragma unroll
    for (int u = 0; u < 8; ++u) S += __expf(acc[u][q] - M);
    rowred[rbase + q][slot] = S;
  }
  __syncthreads();
  if (tid < 16) {
    float S = 0.f;
    for (int s = 0; s < 128; ++s) S += rowred[tid][s];
    rowstat[tid] = rowstat[tid] + __logf(S); // now = row lse
  }
  __syncthreads();
#pragma unroll
  for (int q = 0; q < 8; ++q) {
    float lse = rowstat[rbase + q];
#pragma unroll
    for (int u = 0; u < 8; ++u) {
      const int j = (wave * 8 + u) * 16 + bcol;
      TexpT[(size_t)j * CN + i0 + rbase + q] = (bf16)__expf(acc[u][q] - lse);
    }
  }
}

// ---------- emission normalizers: lse[c] = logsumexp_v(ft[c] . term_emb[v]) ----------
__global__ void k_lse(const bf16* __restrict__ ft_bf, const bf16* __restrict__ term_bf,
                      float* __restrict__ lse) {
  __shared__ float LM[8][32][8];
  __shared__ float LS[8][32][8];
  const int tid = threadIdx.x, lane = tid & 31, wave = tid >> 5;
  const int c0 = blockIdx.x * 16;
  const int aoff = (lane & 16) ? 16 : 0;
  const int boff = (lane & 16) ? 16 : 0;
  const bf16* arow = ft_bf + (size_t)(c0 + (lane & 15)) * HN;
  float m[8], s[8];
#pragma unroll
  for (int q = 0; q < 8; ++q) { m[q] = NEG_INF; s[q] = 0.f; }
  for (int vt = wave; vt < VN / 16; vt += 8) {
    const int v0 = vt * 16;
    const bf16* brow = term_bf + (size_t)(v0 + (lane & 15)) * HN + boff;
    v8f acc = {};
    for (int k0 = 0; k0 < HN; k0 += 32) {
      v16bf a = ld_a(arow + k0 + aoff);
      v16bf b = ld_b(brow + k0);
      acc = wmma_bf16(a, b, acc);
    }
#pragma unroll
    for (int q = 0; q < 8; ++q) {
      float l = acc[q];
      float M = fmaxf(m[q], l);
      s[q] = s[q] * __expf(m[q] - M) + __expf(l - M);
      m[q] = M;
    }
  }
#pragma unroll
  for (int q = 0; q < 8; ++q) { LM[wave][lane][q] = m[q]; LS[wave][lane][q] = s[q]; }
  __syncthreads();
  if (tid < 16) {
    const int q = tid & 7;
    const int l0 = (tid < 8) ? 0 : 16;
    float M = NEG_INF, S = 0.f;
    for (int w = 0; w < 8; ++w)
      for (int li = 0; li < 16; ++li) {
        float mm = LM[w][l0 + li][q], ss = LS[w][l0 + li][q];
        if (mm > M) { S = S * __expf(M - mm) + ss; M = mm; }
        else        { S = S + ss * __expf(mm - M); }
      }
    lse[c0 + tid] = M + __logf(S);
  }
}

// ---------- gather token embeddings to bf16 [p = t*8+n][H] ----------
__global__ void k_etok(const int* __restrict__ text, const bf16* __restrict__ term_bf,
                       bf16* __restrict__ Etok) {
  const int idx = blockIdx.x * 256 + threadIdx.x;
  const int p = idx >> 8, k = idx & 255;
  const int n = p & 7, t = p >> 3;
  const int tok = text[n * TSTEPS + t];
  Etok[(size_t)p * HN + k] = term_bf[(size_t)tok * HN + k];
}

// ---------- gathered emission log-probs: Eg[p][c] = ft[c].Etok[p] - lse[c] ----------
__global__ void k_emis(const bf16* __restrict__ Etok, const bf16* __restrict__ ft_bf,
                       const float* __restrict__ lse, float* __restrict__ Eg) {
  const int tid = threadIdx.x, lane = tid & 31, wave = tid >> 5;
  const int tile = blockIdx.x * 8 + wave;
  const int p0 = (tile >> 6) * 16, c0 = (tile & 63) * 16;
  const int aoff = (lane & 16) ? 16 : 0, boff = (lane & 16) ? 16 : 0;
  const bf16* arow = Etok + (size_t)(p0 + (lane & 15)) * HN;
  const bf16* brow = ft_bf + (size_t)(c0 + (lane & 15)) * HN + boff;
  v8f acc = {};
  for (int k0 = 0; k0 < HN; k0 += 32) {
    v16bf a = ld_a(arow + k0 + aoff);
    v16bf b = ld_b(brow + k0);
    acc = wmma_bf16(a, b, acc);
  }
  const int rbase = (lane & 16) ? 8 : 0;
  const int c = c0 + (lane & 15);
  const float L = lse[c];
#pragma unroll
  for (int q = 0; q < 8; ++q)
    Eg[(size_t)(p0 + rbase + q) * CN + c] = acc[q] - L;
}

// ---------- persistent scan kernel with grid barrier ----------
__device__ __forceinline__ void grid_bar(unsigned* counter, unsigned target) {
  __syncthreads();
  if (threadIdx.x == 0) {
    __hip_atomic_fetch_add(counter, 1u, __ATOMIC_ACQ_REL, __HIP_MEMORY_SCOPE_AGENT);
    while (__hip_atomic_load(counter, __ATOMIC_ACQUIRE, __HIP_MEMORY_SCOPE_AGENT) < target)
      __builtin_amdgcn_s_sleep(1);
  }
  __syncthreads();
}

__global__ void k_scan(const float* __restrict__ startv, const float* __restrict__ Eg,
                       const bf16* __restrict__ TexpT, float* __restrict__ alphaA,
                       float* __restrict__ alphaB, unsigned* counter,
                       float* __restrict__ out) {
  __shared__ bf16 P[16][LDP];
  __shared__ float pm[8][16];
  __shared__ float m_sh[8];
  __shared__ float ls[8];
  extern __shared__ __align__(16) bf16 Tlds[]; // 64 x LDP bf16 = 129 KB (dynamic)
  const int tid = threadIdx.x, lane = tid & 31, wave = tid >> 5;
  const int wg = blockIdx.x;
  // stage this WG's 64-column slice of TexpT into LDS via async DMA
  // (GLOBAL_LOAD_ASYNC_TO_LDS_B128, tracked by ASYNCcnt; generic LDS pointer's
  // low 32 bits are the LDS byte address per ISA 10.2 aperture rules)
  {
    const unsigned tbase = (unsigned)(uintptr_t)(void*)Tlds;
    for (int idx = tid; idx < 64 * (CN / 8); idx += 128) {
      const int r = idx >> 7, o = idx & 127;
      async_g2l_b128(tbase + (unsigned)(r * LDP + o * 8) * 2u,
                     TexpT + (size_t)(wg * 64 + r) * CN + o * 8);
    }
    wait_async0();
  }
  // zero pad rows 8..15 of P once (A-matrix M padding for N=8)
  for (int idx = tid; idx < 8 * LDP; idx += 128)
    P[8 + idx / LDP][idx % LDP] = (bf16)0.f;
  // alpha0 slice: start + emission of token at t=0
  for (int idx = tid; idx < 8 * 64; idx += 128) {
    const int n = idx >> 6;
    const int j = wg * 64 + (idx & 63);
    alphaA[n * CN + j] = startv[j] + Eg[(size_t)n * CN + j];
  }
  grid_bar(counter, SCAN_WGS);
  float* cur = alphaA;
  float* nxt = alphaB;
  const int r8 = tid >> 4;  // 0..7
  const int c16 = tid & 15;
  const int arow = lane & 15;
  const int aoff = (lane & 16) ? 16 : 0;
  const int j0 = wg * 64 + wave * 16;
  const bf16* brow = Tlds + (size_t)(wave * 16 + (lane & 15)) * LDP + ((lane & 16) ? 16 : 0);
  for (int t = 1; t < TSTEPS; ++t) {
    // prefetch next step's emission rows for this WG's column slice
    if (t + 1 < TSTEPS && tid < 8)
      __builtin_prefetch(Eg + (size_t)(t + 1) * 8 * CN + (size_t)tid * CN + wg * 64, 0, 1);
    { // per-batch-row max of alpha
      const float* row = cur + r8 * CN + c16 * 64;
      float M = NEG_INF;
      for (int j = 0; j < 64; ++j) M = fmaxf(M, row[j]);
      pm[r8][c16] = M;
    }
    __syncthreads();
    if (tid < 8) {
      float M = NEG_INF;
      for (int q = 0; q < 16; ++q) M = fmaxf(M, pm[tid][q]);
      m_sh[tid] = M;
    }
    __syncthreads();
    for (int idx = tid; idx < 8 * CN; idx += 128) {
      const int n = idx >> 10, k = idx & 1023;
      P[n][k] = (bf16)__expf(cur[n * CN + k] - m_sh[n]);
    }
    __syncthreads();
    v8f acc = {};
#pragma unroll 4
    for (int k0 = 0; k0 < CN; k0 += 32) {
      v16bf a = ld_a(&P[arow][k0 + aoff]);
      v16bf b = ld_b(brow + k0);
      acc = wmma_bf16(a, b, acc);
    }
    if (lane < 16) { // D rows 0..7 (valid batch rows) live in lanes 0..15
      const int j = j0 + lane;
      const float* eg = Eg + (size_t)t * 8 * CN + j;
#pragma unroll
      for (int q = 0; q < 8; ++q)
        nxt[q * CN + j] = m_sh[q] + __logf(fmaxf(acc[q], 1e-37f)) + eg[(size_t)q * CN];
    }
    grid_bar(counter, (unsigned)(t + 1) * SCAN_WGS);
    float* tp = cur; cur = nxt; nxt = tp;
  }
  if (wg == 0) { // evidence = sum_n logsumexp_j alpha_T[n][j]
    const float* row = cur + r8 * CN + c16 * 64;
    float M = NEG_INF;
    for (int j = 0; j < 64; ++j) M = fmaxf(M, row[j]);
    pm[r8][c16] = M;
    __syncthreads();
    if (tid < 8) {
      float MM = NEG_INF;
      for (int q = 0; q < 16; ++q) MM = fmaxf(MM, pm[tid][q]);
      m_sh[tid] = MM;
    }
    __syncthreads();
    float S = 0.f;
    for (int j = 0; j < 64; ++j) S += __expf(row[j] - m_sh[r8]);
    pm[r8][c16] = S;
    __syncthreads();
    if (tid < 8) {
      float SS = 0.f;
      for (int q = 0; q < 16; ++q) SS += pm[tid][q];
      ls[tid] = m_sh[tid] + __logf(SS);
    }
    __syncthreads();
    if (tid == 0) {
      float e = 0.f;
      for (int n = 0; n < 8; ++n) e += ls[n];
      out[0] = e;
    }
  }
}

extern "C" void kernel_launch(void* const* d_in, const int* in_sizes, int n_in,
                              void* d_out, int out_size, void* d_ws, size_t ws_size,
                              hipStream_t stream) {
  (void)in_sizes; (void)n_in; (void)out_size; (void)ws_size;
  const int*   text         = (const int*)d_in[0];
  const float* start_emb    = (const float*)d_in[1];
  const float* start_lin_W  = (const float*)d_in[2];
  const float* start_lin_b  = (const float*)d_in[3];
  const float* start_res_W1 = (const float*)d_in[4];
  const float* start_res_b1 = (const float*)d_in[5];
  const float* start_res_W2 = (const float*)d_in[6];
  const float* start_res_b2 = (const float*)d_in[7];
  const float* state_emb    = (const float*)d_in[8];
  const float* next_emb     = (const float*)d_in[9];
  const float* pre_emb      = (const float*)d_in[10];
  const float* term_res_W1  = (const float*)d_in[11];
  const float* term_res_b1  = (const float*)d_in[12];
  const float* term_res_W2  = (const float*)d_in[13];
  const float* term_res_b2  = (const float*)d_in[14];
  const float* term_emb     = (const float*)d_in[15];

  char* base = (char*)d_ws;
  unsigned* counter = (unsigned*)base;                       // 4 B (256 reserved)
  float* startv  = (float*)(base + 256);                     // 4 KB
  float* lse     = (float*)(base + 256 + 4096);              // 4 KB
  bf16*  ft_bf   = (bf16*)(base + 256 + 8192);               // 512 KB
  bf16*  TexpT   = ft_bf + (size_t)CN * HN;                  // 2 MB
  bf16*  Etok    = TexpT + (size_t)CN * CN;                  // 2 MB
  float* Eg      = (float*)(Etok + (size_t)NT * HN);         // 16 MB
  float* alphaA  = Eg + (size_t)NT * CN;                     // 32 KB
  float* alphaB  = alphaA + NB * CN;                         // 32 KB
  bf16*  term_bf = (bf16*)(alphaB + NB * CN);                // 16 MB

  hipLaunchKernelGGL(k_init, dim3(1), dim3(1), 0, stream, counter);
  hipLaunchKernelGGL(k_cvt, dim3(VN * HN / 256), dim3(256), 0, stream, term_emb,
                     term_bf);
  hipLaunchKernelGGL(k_start, dim3(1), dim3(256), 0, stream, start_emb, start_lin_W,
                     start_lin_b, start_res_W1, start_res_b1, start_res_W2,
                     start_res_b2, next_emb, startv);
  hipLaunchKernelGGL(k_termmlp, dim3(CN), dim3(256), 0, stream, pre_emb, term_res_W1,
                     term_res_b1, term_res_W2, term_res_b2, ft_bf);
  hipLaunchKernelGGL(k_transition, dim3(CN / 16), dim3(256), 0, stream, state_emb,
                     next_emb, TexpT);
  hipLaunchKernelGGL(k_lse, dim3(CN / 16), dim3(256), 0, stream, ft_bf, term_bf, lse);
  hipLaunchKernelGGL(k_etok, dim3(NT * HN / 256), dim3(256), 0, stream, text, term_bf,
                     Etok);
  hipLaunchKernelGGL(k_emis, dim3((NT / 16) * (CN / 16) / 8), dim3(256), 0, stream,
                     Etok, ft_bf, lse, Eg);
  hipLaunchKernelGGL(k_scan, dim3(SCAN_WGS), dim3(128), 64 * LDP * (int)sizeof(bf16),
                     stream, startv, Eg, TexpT, alphaA, alphaB, counter,
                     (float*)d_out);
}
